// TinyMoE_35966056136993
// MI455X (gfx1250) — matrix operand order repaired and checked by reading, not compiled, not run on hardware
//
#include <hip/hip_runtime.h>
#include <math.h>

#define TB 4
#define TS 2048
#define T_ (TB*TS)      // 8192 tokens
#define H_ 1024
#define I_ 512
#define E_ 16

#define PROW 80          // LDS panel row stride in bytes (64B data + 16B pad)

typedef __bf16 bf16_t;
typedef __attribute__((ext_vector_type(16))) __bf16 bf16x16;
typedef __attribute__((ext_vector_type(8)))  float  f32x8;

union AFrag { bf16x16 v; uint4 q[2]; };

__device__ __forceinline__ f32x8 wmma_bf16(bf16x16 a, bf16x16 b, f32x8 c) {
  return __builtin_amdgcn_wmma_f32_16x16x32_bf16(false, a, false, b, (short)0, c,
                                                 false, false);
}
__device__ __forceinline__ f32x8 zero8() {
  f32x8 z = {0.f,0.f,0.f,0.f,0.f,0.f,0.f,0.f}; return z;
}
// A fragment: 16x32 bf16 tile, row m = lane&15; lanes 0-15 hold K {0..7,16..23},
// lanes 16-31 hold K {8..15,24..31} (ISA 16-bit A layout). Two 16B global loads.
__device__ __forceinline__ void load_a(AFrag& a, const bf16_t* row, int k0, int hi) {
  a.q[0] = *(const uint4*)(row + k0 + hi*8);
  a.q[1] = *(const uint4*)(row + k0 + 16 + hi*8);
}
// B fragment from an LDS panel (64 rows x 32 k, row stride PROW bytes):
// lane holds column n = row, K {8h..8h+7, 16+8h..16+8h+7} with h = (lane>=16).
__device__ __forceinline__ bf16x16 lds_b(const char* p, int row, int hi) {
  union { bf16x16 v; uint4 q[2]; } u;
  const uint4* s = (const uint4*)(p + row*PROW + hi*16);
  u.q[0] = s[0]; u.q[1] = s[2];
  return u.v;
}
__device__ __forceinline__ unsigned lds_off(const void* p) {
  // generic LDS address: bits[31:0] are the in-allocation byte offset
  return (unsigned)(unsigned long long)(uintptr_t)p;
}
// Cooperatively stage a 64xN-row, 32-K bf16 panel into LDS with async loads.
// 64 rows x 64B = 256 chunks of 16B; 128 threads issue 2 chunks each.
__device__ __forceinline__ void async_panel(const bf16_t* __restrict__ gbase, int ldk,
                                            int n0, int k0, unsigned ldsbase, int tid) {
  #pragma unroll
  for (int i = 0; i < 2; ++i) {
    const int chunk = tid + i*128;
    const int row = chunk >> 2;
    const int c4  = chunk & 3;
    const unsigned lds = ldsbase + row*PROW + c4*16;
    const unsigned long long g =
        (unsigned long long)(uintptr_t)(gbase + (size_t)(n0 + row)*ldk + k0 + c4*8);
    asm volatile("global_load_async_to_lds_b128 %0, %1, off"
                 :: "v"(lds), "v"(g) : "memory");
  }
}
__device__ __forceinline__ void wait_async0() {
  asm volatile("s_wait_asynccnt 0x0" ::: "memory");
}

__global__ void zero_i32(int* __restrict__ p, int n) {
  int i = blockIdx.x*blockDim.x + threadIdx.x;
  if (i < n) p[i] = 0;
}

__global__ void cvt_f32_bf16(const float* __restrict__ s, bf16_t* __restrict__ d, int n) {
  int i = blockIdx.x*blockDim.x + threadIdx.x;
  if (i < n) d[i] = (bf16_t)s[i];
}

// one wave per token: 16 expert logits, softmax, top-2, append into expert lists
__global__ __launch_bounds__(256)
void router_topk(const float* __restrict__ x, const float* __restrict__ rw,
                 int* __restrict__ cnt, int* __restrict__ list, float* __restrict__ wgt) {
  __shared__ float sc[8][16];
  const int wave = threadIdx.x >> 5;
  const int lane = threadIdx.x & 31;
  const int tok  = blockIdx.x * 8 + wave;
  const int e    = lane & 15;
  const int half = lane >> 4;
  const float* xr = x  + (size_t)tok * H_ + half * (H_/2);
  const float* wr = rw + (size_t)e   * H_ + half * (H_/2);
  float acc = 0.f;
  #pragma unroll 4
  for (int k = 0; k < H_/2; k += 4) {
    float4 xv = *(const float4*)(xr + k);
    float4 wv = *(const float4*)(wr + k);
    acc += xv.x*wv.x + xv.y*wv.y + xv.z*wv.z + xv.w*wv.w;
  }
  acc += __shfl_xor(acc, 16);
  if (half == 0) sc[wave][e] = acc;
  __syncthreads();
  if (lane == 0) {
    float mx = -1e30f;
    for (int i = 0; i < E_; ++i) mx = fmaxf(mx, sc[wave][i]);
    float sum = 0.f;
    for (int i = 0; i < E_; ++i) sum += __expf(sc[wave][i] - mx);
    int i0 = 0; float b0 = -1e30f;
    for (int i = 0; i < E_; ++i) if (sc[wave][i] > b0) { b0 = sc[wave][i]; i0 = i; }
    int i1 = (i0 == 0) ? 1 : 0; float b1 = -1e30f;
    for (int i = 0; i < E_; ++i) if (i != i0 && sc[wave][i] > b1) { b1 = sc[wave][i]; i1 = i; }
    const float inv = 1.f / sum;
    const int s0 = tok*2, s1 = tok*2 + 1;
    wgt[s0] = __expf(b0 - mx) * inv;
    wgt[s1] = __expf(b1 - mx) * inv;
    int p0 = atomicAdd(cnt + i0, 1); list[i0*T_ + p0] = s0;
    int p1 = atomicAdd(cnt + i1, 1); list[i1*T_ + p1] = s1;
  }
}

// routed gate+up: 64-slot tile (gathered) x 64 of I per block; 4 waves own 16
// rows each and share async-staged, double-buffered B panels in LDS.
__global__ __launch_bounds__(128)
void exp_gateup(const bf16_t* __restrict__ xb, const bf16_t* __restrict__ egw,
                const bf16_t* __restrict__ euw, const int* __restrict__ cnt,
                const int* __restrict__ list, const float* __restrict__ wgt,
                bf16_t* __restrict__ hid) {
  const int e = blockIdx.y;
  const int c = cnt[e];
  const int tile = blockIdx.x;
  if (tile * 64 >= c) return;
  __shared__ int   sslot[64];
  __shared__ float sw[64];
  __shared__ __align__(16) char pg[2][64*PROW];
  __shared__ __align__(16) char pu[2][64*PROW];
  const int tid = threadIdx.x;
  if (tid < 64) {
    int idx = tile*64 + tid;
    int slot = (idx < c) ? list[e*T_ + idx] : -1;
    sslot[tid] = slot;
    sw[tid] = (slot >= 0) ? wgt[slot] : 0.f;
  }
  const int n0 = blockIdx.z * 64;
  const bf16_t* G = egw + (size_t)e * I_ * H_;
  const bf16_t* U = euw + (size_t)e * I_ * H_;
  async_panel(G, H_, n0, 0, lds_off(pg[0]), tid);
  async_panel(U, H_, n0, 0, lds_off(pu[0]), tid);
  __syncthreads();                       // publish sslot/sw (panels still in flight)
  const int lane = tid & 31, wave = tid >> 5;
  const int m = lane & 15, hi = lane >> 4;
  const int mrow = wave*16 + m;
  int slotm = sslot[mrow]; if (slotm < 0) slotm = sslot[0];
  const bf16_t* arow = xb + (size_t)(slotm >> 1) * H_;
  f32x8 accg[4], accu[4];
  #pragma unroll
  for (int j = 0; j < 4; ++j) { accg[j] = zero8(); accu[j] = zero8(); }
  const int NSTEP = H_ / 32;
  for (int s = 0; s < NSTEP; ++s) {
    wait_async0();
    __syncthreads();                     // current panels visible block-wide
    if (s + 1 < NSTEP) {
      async_panel(G, H_, n0, (s+1)*32, lds_off(pg[(s+1)&1]), tid);
      async_panel(U, H_, n0, (s+1)*32, lds_off(pu[(s+1)&1]), tid);
    }
    AFrag a; load_a(a, arow, s*32, hi);
    const char* bg = pg[s&1];
    const char* bu = pu[s&1];
    #pragma unroll
    for (int j = 0; j < 4; ++j) {
      accg[j] = wmma_bf16(a.v, lds_b(bg, j*16 + m, hi), accg[j]);
      accu[j] = wmma_bf16(a.v, lds_b(bu, j*16 + m, hi), accu[j]);
    }
  }
  #pragma unroll
  for (int r = 0; r < 8; ++r) {
    const int mr = wave*16 + r + 8*hi;
    const int slot = sslot[mr];
    if (slot < 0) continue;
    const float w = sw[mr];
    bf16_t* hrow = hid + (size_t)slot * I_;
    #pragma unroll
    for (int j = 0; j < 4; ++j) {
      float g = accg[j][r], u = accu[j][r];
      hrow[n0 + j*16 + m] = (bf16_t)((u * w) / (1.f + __expf(-g)));
    }
  }
}

// routed down proj: hid[slot,I] x exp_down[e][H][I] -> routed[slot,H] (fp32)
__global__ __launch_bounds__(128)
void exp_down(const bf16_t* __restrict__ hid, const bf16_t* __restrict__ edw,
              const int* __restrict__ cnt, const int* __restrict__ list,
              float* __restrict__ routed) {
  const int e = blockIdx.y;
  const int c = cnt[e];
  const int tile = blockIdx.x;
  if (tile * 64 >= c) return;
  __shared__ int sslot[64];
  __shared__ __align__(16) char pd[2][64*PROW];
  const int tid = threadIdx.x;
  if (tid < 64) {
    int idx = tile*64 + tid;
    sslot[tid] = (idx < c) ? list[e*T_ + idx] : -1;
  }
  const int n0 = blockIdx.z * 64;
  const bf16_t* D = edw + (size_t)e * H_ * I_;
  async_panel(D, I_, n0, 0, lds_off(pd[0]), tid);
  __syncthreads();
  const int lane = tid & 31, wave = tid >> 5;
  const int m = lane & 15, hi = lane >> 4;
  const int mrow = wave*16 + m;
  int slotm = sslot[mrow]; if (slotm < 0) slotm = sslot[0];
  const bf16_t* arow = hid + (size_t)slotm * I_;
  f32x8 acc[4];
  #pragma unroll
  for (int j = 0; j < 4; ++j) acc[j] = zero8();
  const int NSTEP = I_ / 32;
  for (int s = 0; s < NSTEP; ++s) {
    wait_async0();
    __syncthreads();
    if (s + 1 < NSTEP) async_panel(D, I_, n0, (s+1)*32, lds_off(pd[(s+1)&1]), tid);
    AFrag a; load_a(a, arow, s*32, hi);
    const char* bd = pd[s&1];
    #pragma unroll
    for (int j = 0; j < 4; ++j)
      acc[j] = wmma_bf16(a.v, lds_b(bd, j*16 + m, hi), acc[j]);
  }
  #pragma unroll
  for (int r = 0; r < 8; ++r) {
    const int mr = wave*16 + r + 8*hi;
    const int slot = sslot[mr];
    if (slot < 0) continue;
    float* orow = routed + (size_t)slot * H_;
    #pragma unroll
    for (int j = 0; j < 4; ++j) orow[n0 + j*16 + m] = acc[j][r];
  }
}

// shared expert gate+up (dense over all tokens)
__global__ __launch_bounds__(128)
void shared_gateup(const bf16_t* __restrict__ xb, const bf16_t* __restrict__ gw,
                   const bf16_t* __restrict__ uw, bf16_t* __restrict__ shh) {
  __shared__ __align__(16) char pg[2][64*PROW];
  __shared__ __align__(16) char pu[2][64*PROW];
  const int tile = blockIdx.x;
  const int n0 = blockIdx.y * 64;
  const int tid = threadIdx.x;
  async_panel(gw, H_, n0, 0, lds_off(pg[0]), tid);
  async_panel(uw, H_, n0, 0, lds_off(pu[0]), tid);
  const int lane = tid & 31, wave = tid >> 5;
  const int m = lane & 15, hi = lane >> 4;
  const bf16_t* arow = xb + (size_t)(tile*64 + wave*16 + m) * H_;
  f32x8 accg[4], accu[4];
  #pragma unroll
  for (int j = 0; j < 4; ++j) { accg[j] = zero8(); accu[j] = zero8(); }
  const int NSTEP = H_ / 32;
  for (int s = 0; s < NSTEP; ++s) {
    wait_async0();
    __syncthreads();
    if (s + 1 < NSTEP) {
      async_panel(gw, H_, n0, (s+1)*32, lds_off(pg[(s+1)&1]), tid);
      async_panel(uw, H_, n0, (s+1)*32, lds_off(pu[(s+1)&1]), tid);
    }
    AFrag a; load_a(a, arow, s*32, hi);
    const char* bg = pg[s&1];
    const char* bu = pu[s&1];
    #pragma unroll
    for (int j = 0; j < 4; ++j) {
      accg[j] = wmma_bf16(a.v, lds_b(bg, j*16 + m, hi), accg[j]);
      accu[j] = wmma_bf16(a.v, lds_b(bu, j*16 + m, hi), accu[j]);
    }
  }
  #pragma unroll
  for (int r = 0; r < 8; ++r) {
    const int tok = tile*64 + wave*16 + r + 8*hi;
    bf16_t* hrow = shh + (size_t)tok * I_;
    #pragma unroll
    for (int j = 0; j < 4; ++j) {
      float g = accg[j][r], u = accu[j][r];
      hrow[n0 + j*16 + m] = (bf16_t)(u / (1.f + __expf(-g)));
    }
  }
}

// shared expert down proj fused with final combine:
// out = x + shared_down(shh) + routed[slot0] + routed[slot1]
__global__ __launch_bounds__(128)
void shared_down_combine(const bf16_t* __restrict__ shh, const bf16_t* __restrict__ dw,
                         const float* __restrict__ x, const float* __restrict__ routed,
                         float* __restrict__ out) {
  __shared__ __align__(16) char pd[2][64*PROW];
  const int tile = blockIdx.x;
  const int n0 = blockIdx.y * 64;
  const int tid = threadIdx.x;
  async_panel(dw, I_, n0, 0, lds_off(pd[0]), tid);
  const int lane = tid & 31, wave = tid >> 5;
  const int m = lane & 15, hi = lane >> 4;
  const bf16_t* arow = shh + (size_t)(tile*64 + wave*16 + m) * I_;
  f32x8 acc[4];
  #pragma unroll
  for (int j = 0; j < 4; ++j) acc[j] = zero8();
  const int NSTEP = I_ / 32;
  for (int s = 0; s < NSTEP; ++s) {
    wait_async0();
    __syncthreads();
    if (s + 1 < NSTEP) async_panel(dw, I_, n0, (s+1)*32, lds_off(pd[(s+1)&1]), tid);
    AFrag a; load_a(a, arow, s*32, hi);
    const char* bd = pd[s&1];
    #pragma unroll
    for (int j = 0; j < 4; ++j)
      acc[j] = wmma_bf16(a.v, lds_b(bd, j*16 + m, hi), acc[j]);
  }
  #pragma unroll
  for (int r = 0; r < 8; ++r) {
    const int tok = tile*64 + wave*16 + r + 8*hi;
    #pragma unroll
    for (int j = 0; j < 4; ++j) {
      const int n = n0 + j*16 + m;
      const size_t o = (size_t)tok * H_ + n;
      out[o] = x[o] + acc[j][r]
             + routed[(size_t)(2*tok)   * H_ + n]
             + routed[(size_t)(2*tok+1) * H_ + n];
    }
  }
}

extern "C" void kernel_launch(void* const* d_in, const int* in_sizes, int n_in,
                              void* d_out, int out_size, void* d_ws, size_t ws_size,
                              hipStream_t stream) {
  (void)in_sizes; (void)n_in; (void)out_size; (void)ws_size;
  const float* x  = (const float*)d_in[0];
  const float* rw = (const float*)d_in[1];
  const float* sg = (const float*)d_in[2];
  const float* su = (const float*)d_in[3];
  const float* sd = (const float*)d_in[4];
  const float* eg = (const float*)d_in[5];
  const float* eu = (const float*)d_in[6];
  const float* ed = (const float*)d_in[7];
  float* out = (float*)d_out;

  char* ws = (char*)d_ws;
  size_t off = 0;
  auto alloc = [&](size_t bytes) -> void* {
    void* p = ws + off; off += (bytes + 255) & ~(size_t)255; return p;
  };
  bf16_t* xb     = (bf16_t*)alloc((size_t)T_ * H_ * 2);
  bf16_t* sgb    = (bf16_t*)alloc((size_t)I_ * H_ * 2);
  bf16_t* sub    = (bf16_t*)alloc((size_t)I_ * H_ * 2);
  bf16_t* sdb    = (bf16_t*)alloc((size_t)H_ * I_ * 2);
  bf16_t* egb    = (bf16_t*)alloc((size_t)E_ * I_ * H_ * 2);
  bf16_t* eub    = (bf16_t*)alloc((size_t)E_ * I_ * H_ * 2);
  bf16_t* edb    = (bf16_t*)alloc((size_t)E_ * H_ * I_ * 2);
  int*    cnt    = (int*)   alloc(E_ * sizeof(int));
  int*    list   = (int*)   alloc((size_t)E_ * T_ * 4);
  float*  wgt    = (float*) alloc((size_t)2 * T_ * 4);
  bf16_t* hidb   = (bf16_t*)alloc((size_t)2 * T_ * I_ * 2);
  bf16_t* shh    = (bf16_t*)alloc((size_t)T_ * I_ * 2);
  float*  routed = (float*) alloc((size_t)2 * T_ * H_ * 4);

  zero_i32<<<1, 32, 0, stream>>>(cnt, E_);

  const int nTH  = T_ * H_;        // 8,388,608
  const int nIH  = I_ * H_;        //   524,288
  const int nEIH = E_ * I_ * H_;   // 8,388,608
  cvt_f32_bf16<<<(nTH  + 255)/256, 256, 0, stream>>>(x,  xb,  nTH);
  cvt_f32_bf16<<<(nIH  + 255)/256, 256, 0, stream>>>(sg, sgb, nIH);
  cvt_f32_bf16<<<(nIH  + 255)/256, 256, 0, stream>>>(su, sub, nIH);
  cvt_f32_bf16<<<(nIH  + 255)/256, 256, 0, stream>>>(sd, sdb, nIH);
  cvt_f32_bf16<<<(nEIH + 255)/256, 256, 0, stream>>>(eg, egb, nEIH);
  cvt_f32_bf16<<<(nEIH + 255)/256, 256, 0, stream>>>(eu, eub, nEIH);
  cvt_f32_bf16<<<(nEIH + 255)/256, 256, 0, stream>>>(ed, edb, nEIH);

  router_topk<<<T_/8, 256, 0, stream>>>(x, rw, cnt, list, wgt);

  exp_gateup<<<dim3(T_/64, E_, I_/64), 128, 0, stream>>>(xb, egb, eub, cnt, list, wgt, hidb);
  exp_down  <<<dim3(T_/64, E_, H_/64), 128, 0, stream>>>(hidb, edb, cnt, list, routed);

  shared_gateup      <<<dim3(T_/64, I_/64), 128, 0, stream>>>(xb, sgb, sub, shh);
  shared_down_combine<<<dim3(T_/64, H_/64), 128, 0, stream>>>(shh, sdb, x, routed, out);
}
